// Block_17978733101066
// MI455X (gfx1250) — compile-verified, matrix operand order checked by reading, not run on hardware
//
#include <hip/hip_runtime.h>
#include <hip/hip_bf16.h>
#include <math.h>

// ---------------- problem constants ----------------
#define NB   8          // batch
#define CCH  96         // channels
#define HW   3136       // 56*56
#define WID  56
#define GRP  4
#define CG   24         // channels per group
#define CGP  32         // padded channels per group
#define HD   384        // mlp hidden
#define NITER 25
#define PTILES 196      // 3136/16 spatial tiles
#define EPSF 1e-12f

typedef __attribute__((ext_vector_type(16))) _Float16 v16h;
typedef __attribute__((ext_vector_type(8)))  _Float16 v8h;
typedef __attribute__((ext_vector_type(8)))  float    v8f;
typedef __attribute__((ext_vector_type(4)))  unsigned int u32x4;
typedef __attribute__((ext_vector_type(4)))  int i32x4;
typedef __attribute__((ext_vector_type(8)))  int i32x8;

// D = A(16x32 f16) * B(32x16 f16) + C(16x16 f32)
__device__ __forceinline__ v8f wmma_f16(v16h a, v16h b, v8f c) {
  return __builtin_amdgcn_wmma_f32_16x16x32_f16(false, a, false, b, (short)0, c,
                                                false, false);
}

// ---- Tensor Data Mover: 1-D copy global -> LDS (nbytes, 8B granules) ----
// D# group0: count=1, lds_addr, 57b global addr, type=2 ("image")
// D# group1: mask=0, data_size=3 (8B), tensor_dim0 = tile_dim0 = nbytes/8, 1-D tile
__device__ __forceinline__ void tdm_load_to_lds(const void* gsrc, void* lds_dst,
                                                unsigned nbytes) {
  unsigned long long ga = (unsigned long long)gsrc;
  unsigned lds = (unsigned)(unsigned long long)lds_dst;  // low 32b = LDS offset
  unsigned n8 = nbytes >> 3;
  u32x4 g0 = { 1u, lds, (unsigned)ga,
               (unsigned)((ga >> 32) & 0x1ffffffu) | (2u << 30) };
  i32x8 g1 = { (int)(3u << 16),                 // data_size=8B, wg_mask=0
               (int)((n8 & 0xffffu) << 16),     // tensor_dim0[15:0]  @ bits 63:48
               (int)((n8 >> 16) & 0xffffu),     // tensor_dim0[31:16] @ bits 79:64
               (int)((n8 & 0xffffu) << 16),     // tile_dim0 @ bits 127:112
               0,                               // tile_dim1=0 (unused), tile_dim2=0
               (int)n8, 0, 0 };                 // tensor_dim0_stride
  i32x4 gz = { 0, 0, 0, 0 };
#if defined(__clang_major__) && (__clang_major__ >= 23)
  i32x8 gz8 = { 0, 0, 0, 0, 0, 0, 0, 0 };
  __builtin_amdgcn_tensor_load_to_lds(g0, g1, gz, gz, gz8, 0);
#else
  __builtin_amdgcn_tensor_load_to_lds(g0, g1, gz, gz, 0);
#endif
}

// A-fragment: lane L -> row m=L%16, K in {kb..kb+7} u {16+kb..16+kb+7}, kb=8*(L>=16)
__device__ __forceinline__ v16h loadA_rows(const _Float16* base, int stride, int koff) {
  int lane = threadIdx.x & 31;
  int m  = lane & 15;
  int kb = (lane & 16) ? 8 : 0;
  const _Float16* s = base + m * stride + koff;
  v8h lo = *(const v8h*)(s + kb);
  v8h hi = *(const v8h*)(s + 16 + kb);
  v16h a;
#pragma unroll
  for (int i = 0; i < 8; ++i) { a[i] = lo[i]; a[8 + i] = hi[i]; }
  return a;
}

// Conv A-fragment with 3x3 tap shift and zero-pad borders.
// act layout: [pixel][32] f16 (channel-padded per group)
__device__ __forceinline__ v16h loadA_conv(const _Float16* actNG, int p0, int dy, int dx) {
  int lane = threadIdx.x & 31;
  int m  = lane & 15;
  int kb = (lane & 16) ? 8 : 0;
  int p  = p0 + m;
  int y  = p / WID + dy - 1;
  int x  = p % WID + dx - 1;
  v16h a;
  if ((unsigned)y < (unsigned)WID && (unsigned)x < (unsigned)WID) {
    const _Float16* s = actNG + (size_t)(y * WID + x) * CGP;
    v8h lo = *(const v8h*)(s + kb);
    v8h hi = *(const v8h*)(s + 16 + kb);
#pragma unroll
    for (int i = 0; i < 8; ++i) { a[i] = lo[i]; a[8 + i] = hi[i]; }
  } else {
#pragma unroll
    for (int i = 0; i < 16; ++i) a[i] = (_Float16)0;
  }
  return a;
}

// B-fragment: prepacked [32 lanes][16 halfs], 32B aligned per lane
__device__ __forceinline__ v16h loadB(const _Float16* frag) {
  int lane = threadIdx.x & 31;
  return *(const v16h*)(frag + lane * 16);
}

// ---------------- weight prep ----------------
__global__ void k_wnorm(const float* __restrict__ w, float* __restrict__ wn) {
  int oc = blockIdx.x * blockDim.x + threadIdx.x;
  if (oc >= CCH) return;
  const float* r = w + oc * (CG * 9);
  float s = 0.f;
  for (int j = 0; j < CG * 9; ++j) s += fabsf(r[j]);
  float inv = 1.f / (s + EPSF);
  float* o = wn + oc * (CG * 9);
  for (int j = 0; j < CG * 9; ++j) o[j] = fabsf(r[j]) * inv;
}

// pack conv B fragments. frag id f = g*18 + tap*2 + nt ; dir0 = forward, dir1 = transpose
__global__ void k_pack_conv(const float* __restrict__ wn,
                            _Float16* __restrict__ Bf, _Float16* __restrict__ Bt) {
  int t = blockIdx.x * blockDim.x + threadIdx.x;
  if (t >= 2 * GRP * 9 * 2 * 32) return;
  int lane = t & 31;
  int fid  = t >> 5;                 // 0..143
  int dir  = fid / 72;               // 0 fwd, 1 transpose
  int f    = fid % 72;
  int g    = f / 18;
  int tap  = (f % 18) / 2;
  int nt   = f & 1;
  int nl   = lane & 15;
  int kb   = (lane & 16) ? 8 : 0;
  _Float16* dst = (dir ? Bt : Bf) + (size_t)f * 512 + lane * 16;
#pragma unroll
  for (int e = 0; e < 16; ++e) {
    int k = (e < 8) ? (kb + e) : (16 + kb + (e - 8));   // K index 0..31
    float v = 0.f;
    if (dir == 0) {                  // forward conv: k = ic, n = oc
      int oc = nt * 16 + nl, ic = k;
      if (oc < CG && ic < CG) v = wn[(g * CG + oc) * (CG * 9) + ic * 9 + tap];
    } else {                         // transpose conv: k = o, n = i, flipped tap
      int ich = nt * 16 + nl, o = k;
      if (ich < CG && o < CG) v = wn[(g * CG + o) * (CG * 9) + ich * 9 + (8 - tap)];
    }
    dst[e] = (_Float16)v;
  }
}

// pack MLP GEMM fragments: w1 (96x384): fid=kc*24+nt ; w2 (384x96): fid-72 = kc*6+nt
__global__ void k_pack_mlp(const float* __restrict__ w1, const float* __restrict__ w2,
                           _Float16* __restrict__ w1p, _Float16* __restrict__ w2p) {
  int t = blockIdx.x * blockDim.x + threadIdx.x;
  if (t >= 144 * 32) return;
  int lane = t & 31;
  int fid  = t >> 5;
  int nl   = lane & 15;
  int kb   = (lane & 16) ? 8 : 0;
  if (fid < 72) {
    int kc = fid / 24, nt = fid % 24;
    _Float16* dst = w1p + (size_t)fid * 512 + lane * 16;
#pragma unroll
    for (int e = 0; e < 16; ++e) {
      int k = (e < 8) ? (kb + e) : (16 + kb + (e - 8));
      dst[e] = (_Float16)w1[(kc * 32 + k) * HD + nt * 16 + nl];
    }
  } else {
    int f2 = fid - 72;
    int kc = f2 / 6, nt = f2 % 6;
    _Float16* dst = w2p + (size_t)f2 * 512 + lane * 16;
#pragma unroll
    for (int e = 0; e < 16; ++e) {
      int k = (e < 8) ? (kb + e) : (16 + kb + (e - 8));
      dst[e] = (_Float16)w2[(kc * 32 + k) * CCH + nt * 16 + nl];
    }
  }
}

// ---------------- state init ----------------
__global__ void k_init(_Float16* __restrict__ h_act, _Float16* __restrict__ r_act,
                       float* __restrict__ hf) {
  int i = blockIdx.x * blockDim.x + threadIdx.x;
  if (i >= NB * GRP * HW * CGP) return;
  int ch = i & (CGP - 1);
  h_act[i] = (ch < CG) ? (_Float16)(1.0f / 96.0f) : (_Float16)0;
  r_act[i] = (_Float16)0;
  if (i < NB * CCH * HW) hf[i] = 1.0f / 96.0f;
}

// channels-first LN + relu + L1 channel-normalize -> xs (NCHW f32)
__global__ void k_prep(const float* __restrict__ x, const float* __restrict__ lw,
                       const float* __restrict__ lb, float* __restrict__ xs) {
  int t = blockIdx.x * blockDim.x + threadIdx.x;
  if (t >= NB * HW) return;
  int n = t / HW, p = t % HW;
  const float* bx = x + (size_t)n * CCH * HW + p;
  float mu = 0.f;
  for (int c = 0; c < CCH; ++c) mu += bx[c * HW];
  mu *= (1.0f / CCH);
  float var = 0.f;
  for (int c = 0; c < CCH; ++c) { float d = bx[c * HW] - mu; var += d * d; }
  var *= (1.0f / CCH);
  float inv = rsqrtf(var + 1e-6f);
  float rsum = 0.f;
  for (int c = 0; c < CCH; ++c) {
    float v = (bx[c * HW] - mu) * inv * lw[c] + lb[c];
    rsum += fmaxf(v, 0.f);
  }
  float inv2 = 1.f / (rsum + EPSF);
  for (int c = 0; c < CCH; ++c) {
    float v = (bx[c * HW] - mu) * inv * lw[c] + lb[c];
    xs[(size_t)(n * CCH + c) * HW + p] = fmaxf(v, 0.f) * inv2;
  }
}

// recon = convT(h); ratio = xs/(recon+eps) stored as f16 act.
// 4 waves/block, all on the same (n,g): TDM stages the group's 18 B-fragments in LDS.
__global__ void k_convt_ratio(const _Float16* __restrict__ h_act,
                              const _Float16* __restrict__ Bt,
                              const float* __restrict__ xs,
                              _Float16* __restrict__ r_act) {
  __shared__ __align__(32) _Float16 sB[18 * 512];   // 18 KB
  int wv  = threadIdx.x >> 5;
  int wid = blockIdx.x * 4 + wv;
  int pt = wid % PTILES;
  int g  = (wid / PTILES) & 3;
  int n  = wid / (PTILES * GRP);
  if (wv == 0) {
    tdm_load_to_lds(Bt + (size_t)g * 18 * 512, sB, 18 * 512 * 2);
    __builtin_amdgcn_s_wait_tensorcnt(0);
  }
  __syncthreads();
  int p0 = pt * 16;
  int ng = n * GRP + g;
  const _Float16* actNG = h_act + (size_t)ng * HW * CGP;
  v8f z = {};
  v8f acc[2]; acc[0] = z; acc[1] = z;
  for (int tap = 0; tap < 9; ++tap) {
    v16h a = loadA_conv(actNG, p0, tap / 3, tap % 3);
    const _Float16* bp = sB + (size_t)(tap * 2) * 512;
    acc[0] = wmma_f16(a, loadB(bp), acc[0]);
    acc[1] = wmma_f16(a, loadB(bp + 512), acc[1]);
  }
  int lane = threadIdx.x & 31;
  int nl = lane & 15;
  int mo = (lane & 16) ? 8 : 0;
#pragma unroll
  for (int nt = 0; nt < 2; ++nt) {
    int ch = nt * 16 + nl;
    if (ch >= CG) continue;
    const float* xsc = xs + (size_t)(n * CCH + g * CG + ch) * HW + p0;
#pragma unroll
    for (int r = 0; r < 8; ++r) {
      int m = r + mo;
      float ratio = xsc[m] / (acc[nt][r] + EPSF);
      ratio = fminf(ratio, 60000.0f);         // keep inside f16 range
      r_act[((size_t)ng * HW + p0 + m) * CGP + ch] = (_Float16)ratio;
    }
  }
}

// mult = conv(ratio); h2 = h_old * mult  (NCHW f32). Same TDM staging.
__global__ void k_conv_mul(const _Float16* __restrict__ r_act,
                           const _Float16* __restrict__ Bf,
                           const float* __restrict__ hf,
                           float* __restrict__ h2) {
  __shared__ __align__(32) _Float16 sB[18 * 512];
  int wv  = threadIdx.x >> 5;
  int wid = blockIdx.x * 4 + wv;
  int pt = wid % PTILES;
  int g  = (wid / PTILES) & 3;
  int n  = wid / (PTILES * GRP);
  if (wv == 0) {
    tdm_load_to_lds(Bf + (size_t)g * 18 * 512, sB, 18 * 512 * 2);
    __builtin_amdgcn_s_wait_tensorcnt(0);
  }
  __syncthreads();
  int p0 = pt * 16;
  int ng = n * GRP + g;
  const _Float16* actNG = r_act + (size_t)ng * HW * CGP;
  v8f z = {};
  v8f acc[2]; acc[0] = z; acc[1] = z;
  for (int tap = 0; tap < 9; ++tap) {
    v16h a = loadA_conv(actNG, p0, tap / 3, tap % 3);
    const _Float16* bp = sB + (size_t)(tap * 2) * 512;
    acc[0] = wmma_f16(a, loadB(bp), acc[0]);
    acc[1] = wmma_f16(a, loadB(bp + 512), acc[1]);
  }
  int lane = threadIdx.x & 31;
  int nl = lane & 15;
  int mo = (lane & 16) ? 8 : 0;
#pragma unroll
  for (int nt = 0; nt < 2; ++nt) {
    int ch = nt * 16 + nl;
    if (ch >= CG) continue;
#pragma unroll
    for (int r = 0; r < 8; ++r) {
      int m = r + mo;
      size_t idx = (size_t)(n * CCH + g * CG + ch) * HW + p0 + m;
      h2[idx] = hf[idx] * acc[nt][r];
    }
  }
}

// channel L1 normalize; refresh f32 h and f16 act copy
__global__ void k_cnorm(const float* __restrict__ h2, float* __restrict__ hf,
                        _Float16* __restrict__ h_act) {
  int t = blockIdx.x * blockDim.x + threadIdx.x;
  if (t >= NB * HW) return;
  int n = t / HW, p = t % HW;
  const float* b = h2 + (size_t)n * CCH * HW + p;
  float s = 0.f;
  for (int c = 0; c < CCH; ++c) s += b[c * HW];
  float inv = 1.f / (s + EPSF);
  for (int c = 0; c < CCH; ++c) {
    float v = b[c * HW] * inv;
    hf[(size_t)(n * CCH + c) * HW + p] = v;
    h_act[((size_t)(n * GRP + c / CG) * HW + p) * CGP + (c % CG)] = (_Float16)v;
  }
}

__global__ void k_resid(const float* __restrict__ x, const float* __restrict__ hf,
                        float* __restrict__ out) {
  int i = blockIdx.x * blockDim.x + threadIdx.x;
  if (i < NB * CCH * HW) out[i] = x[i] + hf[i];
}

// fused LN + (96->384) GELU + (384->96) + residual; one wave per 16 pixels
__global__ void k_mlp(const float* __restrict__ xres, float* __restrict__ out,
                      const _Float16* __restrict__ w1p, const float* __restrict__ b1,
                      const _Float16* __restrict__ w2p, const float* __restrict__ b2,
                      const float* __restrict__ lw, const float* __restrict__ lb) {
  __shared__ __align__(32) _Float16 smem[2 * (16 * CCH + 16 * HD)];
  int wv   = threadIdx.x >> 5;
  _Float16* xln = smem + (size_t)wv * (16 * CCH + 16 * HD);
  _Float16* h1  = xln + 16 * CCH;
  int wid  = blockIdx.x * 2 + wv;
  int pix0 = wid * 16;
  int n = pix0 / HW, pbase = pix0 % HW;   // HW % 16 == 0 -> tile stays in one image
  int lane = threadIdx.x & 31;
  // LayerNorm over channels (lanes 0..15, one pixel each)
  if (lane < 16) {
    const float* bx = xres + (size_t)n * CCH * HW + pbase + lane;
    float mu = 0.f;
    for (int c = 0; c < CCH; ++c) mu += bx[c * HW];
    mu *= (1.0f / CCH);
    float var = 0.f;
    for (int c = 0; c < CCH; ++c) { float d = bx[c * HW] - mu; var += d * d; }
    var *= (1.0f / CCH);
    float inv = rsqrtf(var + 1e-5f);
    for (int c = 0; c < CCH; ++c) {
      float v = (bx[c * HW] - mu) * inv * lw[c] + lb[c];
      xln[lane * CCH + c] = (_Float16)v;
    }
  }
  __syncthreads();
  int nl = lane & 15;
  int mo = (lane & 16) ? 8 : 0;
  // GEMM1 + exact GELU
  for (int nt = 0; nt < HD / 16; ++nt) {
    v8f acc = {};
    for (int kc = 0; kc < CCH / 32; ++kc) {
      v16h a = loadA_rows(xln, CCH, kc * 32);
      v16h b = loadB(w1p + (size_t)(kc * 24 + nt) * 512);
      acc = wmma_f16(a, b, acc);
    }
    int ch = nt * 16 + nl;
    float bias = b1[ch];
#pragma unroll
    for (int r = 0; r < 8; ++r) {
      float v = acc[r] + bias;
      float gl = 0.5f * v * (1.f + erff(v * 0.70710678118f));
      h1[(r + mo) * HD + ch] = (_Float16)gl;
    }
  }
  __syncthreads();
  // GEMM2 + residual
  for (int nt = 0; nt < CCH / 16; ++nt) {
    v8f acc = {};
    for (int kc = 0; kc < HD / 32; ++kc) {
      v16h a = loadA_rows(h1, HD, kc * 32);
      v16h b = loadB(w2p + (size_t)(kc * 6 + nt) * 512);
      acc = wmma_f16(a, b, acc);
    }
    int ch = nt * 16 + nl;
    float bias = b2[ch];
#pragma unroll
    for (int r = 0; r < 8; ++r) {
      size_t idx = (size_t)(n * CCH + ch) * HW + pbase + r + mo;
      out[idx] = xres[idx] + acc[r] + bias;
    }
  }
}

// ---------------- launcher ----------------
extern "C" void kernel_launch(void* const* d_in, const int* in_sizes, int n_in,
                              void* d_out, int out_size, void* d_ws, size_t ws_size,
                              hipStream_t stream) {
  const float* x    = (const float*)d_in[0];
  const float* ln1w = (const float*)d_in[1];
  const float* ln1b = (const float*)d_in[2];
  const float* wnn  = (const float*)d_in[3];
  const float* ln2w = (const float*)d_in[4];
  const float* ln2b = (const float*)d_in[5];
  const float* w1   = (const float*)d_in[6];
  const float* b1   = (const float*)d_in[7];
  const float* w2   = (const float*)d_in[8];
  const float* b2   = (const float*)d_in[9];
  float* out = (float*)d_out;

  char* ws = (char*)d_ws;
  size_t off = 0;
  auto alloc = [&](size_t bytes) -> void* {
    off = (off + 255) & ~(size_t)255;
    void* p = ws + off;
    off += bytes;
    return p;
  };
  float*     xs    = (float*)alloc((size_t)NB * CCH * HW * 4);
  float*     hf    = (float*)alloc((size_t)NB * CCH * HW * 4);
  float*     h2    = (float*)alloc((size_t)NB * CCH * HW * 4);
  _Float16*  h_act = (_Float16*)alloc((size_t)NB * GRP * HW * CGP * 2);
  _Float16*  r_act = (_Float16*)alloc((size_t)NB * GRP * HW * CGP * 2);
  float*     wn    = (float*)alloc((size_t)CCH * CG * 9 * 4);
  _Float16*  Bf    = (_Float16*)alloc((size_t)72 * 512 * 2);
  _Float16*  Bt    = (_Float16*)alloc((size_t)72 * 512 * 2);
  _Float16*  w1p   = (_Float16*)alloc((size_t)72 * 512 * 2);
  _Float16*  w2p   = (_Float16*)alloc((size_t)72 * 512 * 2);

  // weight prep + packing
  k_wnorm<<<1, 128, 0, stream>>>(wnn, wn);
  k_pack_conv<<<18, 256, 0, stream>>>(wn, Bf, Bt);
  k_pack_mlp<<<18, 256, 0, stream>>>(w1, w2, w1p, w2p);
  // state init + input prep
  k_init<<<(NB * GRP * HW * CGP + 255) / 256, 256, 0, stream>>>(h_act, r_act, hf);
  k_prep<<<(NB * HW + 255) / 256, 256, 0, stream>>>(x, ln1w, ln1b, xs);
  // NNMF multiplicative updates (4 waves/block, block-uniform (n,g) for LDS B tile)
  int convBlocks = (NB * GRP * PTILES) / 4;   // 1568 blocks x 4 waves
  for (int it = 0; it < NITER; ++it) {
    k_convt_ratio<<<convBlocks, 128, 0, stream>>>(h_act, Bt, xs, r_act);
    k_conv_mul  <<<convBlocks, 128, 0, stream>>>(r_act, Bf, hf, h2);
    k_cnorm     <<<(NB * HW + 255) / 256, 256, 0, stream>>>(h2, hf, h_act);
  }
  // residual into d_out, then fused MLP reads/writes d_out per-pixel
  k_resid<<<(NB * CCH * HW + 255) / 256, 256, 0, stream>>>(x, hf, out);
  k_mlp<<<(NB * HW / 16) / 2, 64, 0, stream>>>(out, out, w1p, b1, w2p, b2, ln2w, ln2b);
}
